// SparseFocusedGroupAttention_83580063580507
// MI455X (gfx1250) — matrix operand order repaired and checked by reading, not compiled, not run on hardware
//
#include <hip/hip_runtime.h>

// ---------------------------------------------------------------------------
// SparseFocusedGroupAttention for MI455X (gfx1250, wave32, WMMA + TDM)
//
// B=1, S=2048, H=4096, HQ=64, G=16, DH=64, R=4, scale=1/8
// Pipeline (bf16 compute, fp32 WMMA accumulation):
//   1) convert x, Wq, Wk, Wv, Wo  fp32 -> bf16 (workspace)
//   2) qb/kb/vb projections via tiled v_wmma_f32_16x16x32_bf16 GEMM
//      (A-tile staged into LDS by the Tensor Data Mover)
//   3) flash attention w/ boolean mask (K-tile + mask tile staged by TDM)
//   4) d_out = ob @ Wob + bo (fp32)
// ---------------------------------------------------------------------------

typedef __attribute__((ext_vector_type(8)))  __bf16 v8bf;
typedef __attribute__((ext_vector_type(16))) __bf16 v16bf;
typedef __attribute__((ext_vector_type(8)))  float  v8f;
typedef __attribute__((ext_vector_type(4)))  unsigned int u32x4;
typedef __attribute__((ext_vector_type(8)))  int i32x8;
typedef __attribute__((ext_vector_type(4)))  int i32x4;

#define SHUF16(a, b) __builtin_shufflevector((a), (b), 0,1,2,3,4,5,6,7,8,9,10,11,12,13,14,15)

#define WMMA_BF16(A, B, C) \
  __builtin_amdgcn_wmma_f32_16x16x32_bf16(false, (A), false, (B), (short)0, (C), false, false)

static constexpr int S  = 2048;
static constexpr int H  = 4096;
static constexpr int GD = 1024;   // G * DH
static constexpr float SCALE = 0.125f;
static constexpr float NEGINF = -1e30f;

// ---------------------------------------------------------------------------
// Tensor Data Mover: 2-D tile load Global -> LDS.
// Descriptor per CDNA5 ISA 8.3/8.4 (D# group0/group1); groups 2/3 zero (2-D).
//   data_size_log2: 0 = 1B elements, 1 = 2B elements
//   dims/strides in data_size units
// Must be executed by a single wave (EXEC ignored; wave-level DMA issue).
// ---------------------------------------------------------------------------
__device__ __forceinline__ void tdm_load_2d(unsigned lds_off, const void* gtile,
                                            unsigned data_size_log2,
                                            unsigned tensor_d0, unsigned tensor_d1,
                                            unsigned tile_d0, unsigned tile_d1,
                                            unsigned stride_d0) {
  const unsigned long long ga = (unsigned long long)gtile;
  u32x4 g0;
  g0[0] = 1u;                                        // count=1, user descriptor
  g0[1] = lds_off;                                   // lds_addr (bytes)
  g0[2] = (unsigned)(ga & 0xffffffffu);              // global_addr[31:0]
  g0[3] = (unsigned)((ga >> 32) & 0x01ffffffu) |     // global_addr[56:32]
          (2u << 30);                                // type = 2 ("image")
  i32x8 g1;
  g1[0] = (int)(data_size_log2 << 16);               // wg_mask=0, data_size
  g1[1] = (int)((tensor_d0 & 0xffffu) << 16);        // tensor_dim0[15:0]
  g1[2] = (int)(((tensor_d0 >> 16) & 0xffffu) |
                ((tensor_d1 & 0xffffu) << 16));      // dim0[31:16] | dim1[15:0]
  g1[3] = (int)(((tensor_d1 >> 16) & 0xffffu) |
                ((tile_d0 & 0xffffu) << 16));        // dim1[31:16] | tile_dim0
  g1[4] = (int)(tile_d1 & 0xffffu);                  // tile_dim1, tile_dim2=0
  g1[5] = (int)stride_d0;                            // tensor_dim0_stride[31:0]
  g1[6] = 0;                                         // stride hi / dim1_stride lo
  g1[7] = 0;                                         // dim1_stride hi
  const i32x4 z4 = {0, 0, 0, 0};
#if __clang_major__ >= 23
  const i32x8 z8 = {0, 0, 0, 0, 0, 0, 0, 0};
  __builtin_amdgcn_tensor_load_to_lds(g0, g1, z4, z4, z8, 0);
#else
  __builtin_amdgcn_tensor_load_to_lds(g0, g1, z4, z4, 0);
#endif
}

// ---------------------------------------------------------------------------
// fp32 -> bf16 conversion (grid-stride)
// ---------------------------------------------------------------------------
__global__ __launch_bounds__(256) void f32_to_bf16_kernel(
    const float* __restrict__ in, __bf16* __restrict__ out, int n) {
  int i = blockIdx.x * blockDim.x + threadIdx.x;
  int stride = gridDim.x * blockDim.x;
  for (; i < n; i += stride) out[i] = (__bf16)in[i];
}

// ---------------------------------------------------------------------------
// GEMM: C[M,N] = A[M,K] @ B[K,N] + bias[N]
// A,B bf16 row-major; OUT_BF16 selects bf16 (intermediate) or fp32 (final) C.
// Block tile 128x128, BK=32, 256 threads (8 waves), wave tile 64x32 (4x2 WMMA)
// A tile staged by TDM; B tile staged cooperatively (needs transpose).
// ---------------------------------------------------------------------------
template <bool OUT_BF16>
__global__ __launch_bounds__(256) void gemm_bf16_wmma(
    const __bf16* __restrict__ A, const __bf16* __restrict__ B,
    const float* __restrict__ bias, void* __restrict__ Cout,
    int M, int N, int K) {
  __shared__ alignas(16) __bf16 sA[128][32];
  __shared__ alignas(16) __bf16 sBt[128][32];   // transposed: [n][k]

  const int tid   = threadIdx.x;
  const int lane  = tid & 31;
  const int w     = tid >> 5;
  const int l15   = lane & 15;
  const int lhalf = lane >> 4;

  const int m0 = blockIdx.y * 128;
  const int n0 = blockIdx.x * 128;
  const int wm = (w & 1) * 64;   // wave row base within tile
  const int wn = (w >> 1) * 32;  // wave col base within tile

  v8f acc[4][2];
  const v8f vzero = {0.f, 0.f, 0.f, 0.f, 0.f, 0.f, 0.f, 0.f};
#pragma unroll
  for (int i = 0; i < 4; ++i)
#pragma unroll
    for (int j = 0; j < 2; ++j) acc[i][j] = vzero;

  const int bk = tid >> 3, bn = (tid & 7) * 16;  // B-tile load map: 32x128
  const unsigned sA_off = (unsigned)(unsigned long long)&sA[0][0];

  for (int k0 = 0; k0 < K; k0 += 32) {
    if (k0 + 32 < K)  // prefetch next B k-tile -> global_prefetch_b8
      __builtin_prefetch(B + (size_t)(k0 + 32 + bk) * N + n0 + bn, 0, 1);

    // stage A tile (128 rows x 32 cols bf16, row stride K) via Tensor DMA
    if (tid < 32) {
      tdm_load_2d(sA_off, A + (size_t)m0 * K + k0, /*2B*/ 1,
                  (unsigned)K, (unsigned)M, /*tile_d0*/ 32, /*tile_d1*/ 128,
                  (unsigned)K);
      __builtin_amdgcn_s_wait_tensorcnt(0);
    }

    // stage B tile transposed (coalesced global read, scattered LDS write)
    const __bf16* gb = B + (size_t)(k0 + bk) * N + n0 + bn;
    v8bf b0 = *(const v8bf*)(gb);
    v8bf b1 = *(const v8bf*)(gb + 8);
#pragma unroll
    for (int i = 0; i < 8; ++i) sBt[bn + i][bk] = b0[i];
#pragma unroll
    for (int i = 0; i < 8; ++i) sBt[bn + 8 + i][bk] = b1[i];
    __syncthreads();

#pragma unroll
    for (int i = 0; i < 4; ++i) {
      // A fragment: lane row = l15; K = lhalf*8 + {0..7} and +16
      const __bf16* pa = &sA[wm + i * 16 + l15][lhalf * 8];
      v16bf afrag = SHUF16(*(const v8bf*)pa, *(const v8bf*)(pa + 16));
#pragma unroll
      for (int j = 0; j < 2; ++j) {
        // B fragment: lane col = l15; K = lhalf*8 + {0..7} and +16
        const __bf16* pb = &sBt[wn + j * 16 + l15][lhalf * 8];
        v16bf bfrag = SHUF16(*(const v8bf*)pb, *(const v8bf*)(pb + 16));
        acc[i][j] = WMMA_BF16(afrag, bfrag, acc[i][j]);
      }
    }
    __syncthreads();
  }

  // epilogue: C layout row = reg + 8*lhalf, col = l15
#pragma unroll
  for (int i = 0; i < 4; ++i)
#pragma unroll
    for (int j = 0; j < 2; ++j)
#pragma unroll
      for (int r = 0; r < 8; ++r) {
        const int row = m0 + wm + i * 16 + r + lhalf * 8;
        const int col = n0 + wn + j * 16 + l15;
        const float v = acc[i][j][r] + bias[col];
        if constexpr (OUT_BF16)
          ((__bf16*)Cout)[(size_t)row * N + col] = (__bf16)v;
        else
          ((float*)Cout)[(size_t)row * N + col] = v;
      }
}

// ---------------------------------------------------------------------------
// Flash attention with boolean keep-mask.
// grid = (S/128, 64 heads); 256 threads (8 waves); wave owns 16 query rows.
// T loop in tiles of 64. Scores & PV via v_wmma_f32_16x16x32_bf16.
// K tile and mask tile staged by TDM; V tile staged cooperatively (transpose).
// ---------------------------------------------------------------------------
__global__ __launch_bounds__(256) void attn_bf16_wmma(
    const __bf16* __restrict__ qb, const __bf16* __restrict__ kb,
    const __bf16* __restrict__ vb, const unsigned char* __restrict__ mask,
    __bf16* __restrict__ ob) {
  __shared__ alignas(16) __bf16 sK[64][64];            // [t][d]
  __shared__ alignas(16) __bf16 sVt[64][64];           // [d][t]
  __shared__ alignas(16) unsigned char sM[128][64];    // [q][t]
  __shared__ alignas(16) __bf16 sP[8][16][64];         // per-wave P tile

  const int tid   = threadIdx.x;
  const int lane  = tid & 31;
  const int w     = tid >> 5;
  const int l15   = lane & 15;
  const int lhalf = lane >> 4;

  const int qbase = blockIdx.x * 128;
  const int h     = blockIdx.y;   // query head 0..63
  const int g     = h >> 2;       // kv group = h / R
  const int qcol  = h * 64;
  const int kcol  = g * 64;

  // Q fragments held in registers for the whole kernel (two 32-wide K chunks)
  const __bf16* qrow =
      qb + (size_t)(qbase + w * 16 + l15) * H + qcol + lhalf * 8;
  v16bf aq[2];
#pragma unroll
  for (int kt = 0; kt < 2; ++kt) {
    v8bf q0 = *(const v8bf*)(qrow + kt * 32);
    v8bf q1 = *(const v8bf*)(qrow + kt * 32 + 16);
    aq[kt] = SHUF16(q0, q1);
  }

  float mrow[8], lrow[8];
  v8f oacc[4];
  const v8f vzero = {0.f, 0.f, 0.f, 0.f, 0.f, 0.f, 0.f, 0.f};
#pragma unroll
  for (int r = 0; r < 8; ++r) { mrow[r] = NEGINF; lrow[r] = 0.f; }
#pragma unroll
  for (int dn = 0; dn < 4; ++dn) oacc[dn] = vzero;

  const int kr = tid >> 2, seg = (tid & 3) * 16;   // V tile load map 64x64
  const unsigned sK_off = (unsigned)(unsigned long long)&sK[0][0];
  const unsigned sM_off = (unsigned)(unsigned long long)&sM[0][0];

  for (int t0 = 0; t0 < S; t0 += 64) {
    // --- stage K tile (64x64 bf16, row stride GD) + mask tile via TDM ---
    if (tid < 32) {
      tdm_load_2d(sK_off, kb + (size_t)t0 * GD + kcol, /*2B*/ 1,
                  GD, S, /*tile_d0*/ 64, /*tile_d1*/ 64, GD);
      tdm_load_2d(sM_off, mask + (size_t)qbase * S + t0, /*1B*/ 0,
                  S, S, /*tile_d0*/ 64, /*tile_d1*/ 128, S);
      __builtin_amdgcn_s_wait_tensorcnt(0);
    }
    // --- stage V tile transposed ([d][t]) cooperatively ---
    const __bf16* gv = vb + (size_t)(t0 + kr) * GD + kcol + seg;
    v8bf v0 = *(const v8bf*)gv;
    v8bf v1 = *(const v8bf*)(gv + 8);
#pragma unroll
    for (int i = 0; i < 8; ++i) sVt[seg + i][kr] = v0[i];
#pragma unroll
    for (int i = 0; i < 8; ++i) sVt[seg + 8 + i][kr] = v1[i];
    __syncthreads();

    // --- scores: S = Q @ K^T ; B-fragment lane col = t, K = d contiguous ---
    v8f sacc[4];
#pragma unroll
    for (int tn = 0; tn < 4; ++tn) sacc[tn] = vzero;
#pragma unroll
    for (int kt = 0; kt < 2; ++kt) {
#pragma unroll
      for (int tn = 0; tn < 4; ++tn) {
        const __bf16* pk = &sK[tn * 16 + l15][kt * 32 + lhalf * 8];
        v16bf bfrag = SHUF16(*(const v8bf*)pk, *(const v8bf*)(pk + 16));
        sacc[tn] = WMMA_BF16(aq[kt], bfrag, sacc[tn]);
      }
    }

    // --- mask + online softmax (row = r + 8*lhalf, cols across 16 lanes) ---
    float preg[4][8];
#pragma unroll
    for (int r = 0; r < 8; ++r) {
      const int rloc = w * 16 + r + lhalf * 8;  // query row within 128 block
      float sv[4];
#pragma unroll
      for (int tn = 0; tn < 4; ++tn) {
        float s = sacc[tn][r] * SCALE;
        if (!sM[rloc][tn * 16 + l15]) s = NEGINF;
        sv[tn] = s;
      }
      float rm = fmaxf(fmaxf(sv[0], sv[1]), fmaxf(sv[2], sv[3]));
#pragma unroll
      for (int off = 1; off <= 8; off <<= 1)
        rm = fmaxf(rm, __shfl_xor(rm, off, 32));   // stays within 16-lane half
      const float mold = mrow[r];
      const float mnew = fmaxf(mold, rm);
      const float alpha = __expf(mold - mnew);
      float psum = 0.f;
#pragma unroll
      for (int tn = 0; tn < 4; ++tn) {
        float p = __expf(sv[tn] - mnew);
        p = (sv[tn] == NEGINF) ? 0.f : p;
        preg[tn][r] = p;
        psum += p;
      }
#pragma unroll
      for (int off = 1; off <= 8; off <<= 1)
        psum += __shfl_xor(psum, off, 32);
      lrow[r] = lrow[r] * alpha + psum;
      mrow[r] = mnew;
#pragma unroll
      for (int dn = 0; dn < 4; ++dn) oacc[dn][r] *= alpha;
    }

    // --- P (C-layout) -> per-wave LDS -> A-fragment layout ---
#pragma unroll
    for (int tn = 0; tn < 4; ++tn)
#pragma unroll
      for (int r = 0; r < 8; ++r)
        sP[w][r + lhalf * 8][tn * 16 + l15] = (__bf16)preg[tn][r];

    // --- PV: O += P @ V ; B-fragment lane col = d, K = t contiguous in sVt ---
#pragma unroll
    for (int kt = 0; kt < 2; ++kt) {
      const __bf16* pp = &sP[w][l15][kt * 32 + lhalf * 8];
      v16bf pfrag = SHUF16(*(const v8bf*)pp, *(const v8bf*)(pp + 16));
#pragma unroll
      for (int dn = 0; dn < 4; ++dn) {
        const __bf16* pv = &sVt[dn * 16 + l15][kt * 32 + lhalf * 8];
        v16bf vfrag = SHUF16(*(const v8bf*)pv, *(const v8bf*)(pv + 16));
        oacc[dn] = WMMA_BF16(pfrag, vfrag, oacc[dn]);
      }
    }
    __syncthreads();
  }

  // --- normalize and write attention output (bf16) ---
#pragma unroll
  for (int dn = 0; dn < 4; ++dn)
#pragma unroll
    for (int r = 0; r < 8; ++r) {
      const int row = qbase + w * 16 + r + lhalf * 8;
      const int col = qcol + dn * 16 + l15;
      const float denom = lrow[r] > 0.f ? lrow[r] : 1.f;
      ob[(size_t)row * H + col] = (__bf16)(oacc[dn][r] / denom);
    }
}

// ---------------------------------------------------------------------------
// Host-side orchestration
// ---------------------------------------------------------------------------
extern "C" void kernel_launch(void* const* d_in, const int* in_sizes, int n_in,
                              void* d_out, int out_size, void* d_ws, size_t ws_size,
                              hipStream_t stream) {
  const float* x  = (const float*)d_in[0];
  const float* Wq = (const float*)d_in[1];
  const float* bq = (const float*)d_in[2];
  const float* Wk = (const float*)d_in[3];
  const float* bk = (const float*)d_in[4];
  const float* Wv = (const float*)d_in[5];
  const float* bv = (const float*)d_in[6];
  const float* Wo = (const float*)d_in[7];
  const float* bo = (const float*)d_in[8];
  const unsigned char* mask = (const unsigned char*)d_in[9];

  char* ws = (char*)d_ws;
  const size_t MB = 1ull << 20;
  __bf16* xb  = (__bf16*)(ws + 0);        // 16 MB  (2048*4096*2)
  __bf16* qb  = (__bf16*)(ws + 16 * MB);  // 16 MB
  __bf16* ob  = (__bf16*)(ws + 32 * MB);  // 16 MB
  __bf16* Wqb = (__bf16*)(ws + 48 * MB);  // 32 MB
  __bf16* Wob = (__bf16*)(ws + 80 * MB);  // 32 MB
  __bf16* Wkb = (__bf16*)(ws + 112 * MB); //  8 MB
  __bf16* Wvb = (__bf16*)(ws + 120 * MB); //  8 MB
  __bf16* kb  = (__bf16*)(ws + 128 * MB); //  4 MB
  __bf16* vb  = (__bf16*)(ws + 132 * MB); //  4 MB  (total 136 MB)

  f32_to_bf16_kernel<<<2048, 256, 0, stream>>>(x, xb, S * H);
  f32_to_bf16_kernel<<<2048, 256, 0, stream>>>(Wq, Wqb, H * H);
  f32_to_bf16_kernel<<<2048, 256, 0, stream>>>(Wk, Wkb, H * GD);
  f32_to_bf16_kernel<<<2048, 256, 0, stream>>>(Wv, Wvb, H * GD);
  f32_to_bf16_kernel<<<2048, 256, 0, stream>>>(Wo, Wob, H * H);

  dim3 blk(256);
  // Q / K / V projections (bf16 outputs)
  gemm_bf16_wmma<true><<<dim3(H / 128, S / 128), blk, 0, stream>>>(
      xb, Wqb, bq, (void*)qb, S, H, H);
  gemm_bf16_wmma<true><<<dim3(GD / 128, S / 128), blk, 0, stream>>>(
      xb, Wkb, bk, (void*)kb, S, GD, H);
  gemm_bf16_wmma<true><<<dim3(GD / 128, S / 128), blk, 0, stream>>>(
      xb, Wvb, bv, (void*)vb, S, GD, H);

  // masked GQA flash attention
  attn_bf16_wmma<<<dim3(S / 128, 64), blk, 0, stream>>>(qb, kb, vb, mask, ob);

  // output projection (fp32 into d_out)
  gemm_bf16_wmma<false><<<dim3(H / 128, S / 128), blk, 0, stream>>>(
      ob, Wob, bo, d_out, S, H, H);
}
